// CCAMemoryModel_52424370815650
// MI455X (gfx1250) — matrix operand order repaired and checked by reading, not compiled, not run on hardware
//
#include <hip/hip_runtime.h>
#include <hip/hip_bf16.h>
#include <math.h>

typedef _Float16 half_t;
typedef __attribute__((ext_vector_type(16))) _Float16 v16h;
typedef __attribute__((ext_vector_type(8)))  float    v8f;

// ---------------------------------------------------------------------------
// Problem constants (from reference):
//  feat_b [8,640,10,10], feat_m [50,640,10,10]
//  embed R=64, corr [8,50,100(ij),100(hw)], conv 1->16->1 separable 4D,
//  att_mem [8,50,640], logits [8,1000]
// ---------------------------------------------------------------------------
#define B_   8
#define N_   50
#define C_   640
#define R_   64
#define HW_  100      // 10x10 spatial
#define NIJ_ 5000     // N_*HW_
#define TEMP_ 5.0f

// ---------------------------------------------------------------------------
// Kernel 1: channel-mean shift + 1x1 conv embed + BN + ReLU + L2 norm -> f16
// One 64-thread block per (image, position).  img 0..7 = feat_b, 8..57 = feat_m
// ---------------------------------------------------------------------------
__global__ __launch_bounds__(64) void embed_kernel(
    const float* __restrict__ feat_b, const float* __restrict__ feat_m,
    const float* __restrict__ w1x1, const float* __restrict__ bn_scale,
    const float* __restrict__ bn_bias,
    half_t* __restrict__ fb16, half_t* __restrict__ fm16)
{
    __shared__ float xs[C_];
    __shared__ float red[64];
    const int blk = blockIdx.x;            // 0..5799
    const int img = blk / HW_;
    const int pos = blk % HW_;
    const int t   = threadIdx.x;           // 0..63

    const float* src = (img < B_) ? (feat_b + (size_t)img * (C_*HW_))
                                  : (feat_m + (size_t)(img - B_) * (C_*HW_));
    // load 10 channels per thread, partial sum for mean
    float vals[10];
    float s = 0.f;
    #pragma unroll
    for (int i = 0; i < 10; ++i) {
        vals[i] = src[(size_t)(t + i*64) * HW_ + pos];
        s += vals[i];
    }
    red[t] = s; __syncthreads();
    #pragma unroll
    for (int st = 32; st > 0; st >>= 1) {
        if (t < st) red[t] += red[t + st];
        __syncthreads();
    }
    const float mean = red[0] * (1.0f / 640.0f);
    #pragma unroll
    for (int i = 0; i < 10; ++i) xs[t + i*64] = vals[i] - mean;
    __syncthreads();

    // output channel o == t : y = sum_c xs[c] * w1x1[o,c]
    const float* wr = w1x1 + (size_t)t * C_;
    float y = 0.f;
    #pragma unroll 4
    for (int c = 0; c < C_; ++c) y += xs[c] * wr[c];
    y = y * bn_scale[t] + bn_bias[t];
    y = fmaxf(y, 0.f);

    red[t] = y * y; __syncthreads();
    #pragma unroll
    for (int st = 32; st > 0; st >>= 1) {
        if (t < st) red[t] += red[t + st];
        __syncthreads();
    }
    const float nrm = fmaxf(sqrtf(red[0]), 1e-8f);
    const half_t hv = (half_t)(y / nrm);

    if (img < B_) fb16[((size_t)img * HW_ + pos) * R_ + t] = hv;
    else          fm16[((size_t)(img - B_) * HW_ + pos) * R_ + t] = hv;
}

// ---------------------------------------------------------------------------
// Kernel 2: correlation GEMM via WMMA.
//   corr[b, (n,ij), hw] = sum_c fb16[b,hw,c] * fm16[(n,ij),c]
// Per batch b: C[M=100, N=5000] = A[100x64] * B^T,  K=64 = 2 x f16 WMMA k32.
// Block = 8 waves: one 16-row M tile, 8 x 16-col tiles.  A staged in LDS.
// Fragment layouts per CDNA5 ISA 7.12.2.  Epilogue: each lane's 8 accumulator
// floats are contiguous rows -> vector v8f store for full tiles, float4 for
// the single partial M tile (rows 96..99, laneHigh==0 only).
// ---------------------------------------------------------------------------
__global__ __launch_bounds__(256) void corr_wmma_kernel(
    const half_t* __restrict__ fb16, const half_t* __restrict__ fm16,
    float* __restrict__ corr)
{
    __shared__ half_t As[16 * 64];
    const int b     = blockIdx.z;
    const int Mbase = blockIdx.y * 16;      // 7 tiles cover 112 >= 100
    const int Nbase = blockIdx.x * 128;     // 40 tiles cover 5120 >= 5000
    const int tid   = threadIdx.x;

    // stage A tile (16 rows x 64 K), zero-pad rows >= 100
    #pragma unroll
    for (int j = 0; j < 4; ++j) {
        const int i  = tid * 4 + j;
        const int ml = i >> 6, k = i & 63;
        const int row = Mbase + ml;
        As[i] = (row < HW_) ? fb16[((size_t)b * HW_ + row) * R_ + k] : (half_t)0.0f;
    }
    __syncthreads();

    const int wave     = tid >> 5;
    const int lane     = tid & 31;
    const int laneHigh = lane >> 4;         // 0/1
    const int lr       = lane & 15;
    const int cg       = Nbase + wave * 16 + lr;          // global column (n*100+ij)
    const int cgc      = (cg < NIJ_) ? cg : (NIJ_ - 1);   // clamp loads

    v8f acc = {};
    #pragma unroll
    for (int chunk = 0; chunk < 2; ++chunk) {
        const int kb = chunk * 32;
        v16h a, bm;
        #pragma unroll
        for (int t = 0; t < 16; ++t) {
            // A 16x32 f16 layout: lanes0-15 K={0..7,16..23}, lanes16-31 K={8..15,24..31}
            const int ka = kb + t + ((t >= 8) ? 8 : 0) + laneHigh * 8;
            a[t] = As[lr * 64 + ka];
            // B 32x16 layout: lane holds col lr, same K order as A
            const int kbK = kb + t + ((t >= 8) ? 8 : 0) + laneHigh * 8;
            bm[t] = fm16[(size_t)cgc * R_ + kbK];
        }
        acc = __builtin_amdgcn_wmma_f32_16x16x32_f16(
                  false, a, false, bm, (short)0, acc, false, false);
    }

    if (cg < NIJ_) {
        float* dst = corr + (size_t)b * (size_t)(NIJ_*HW_)
                          + (size_t)cg * HW_ + (Mbase + 8 * laneHigh);
        if (Mbase + 16 <= HW_) {
            // full tile: 8 consecutive rows per lane, 16B-aligned (32B only for even cg)
            *(v8f*)dst = acc;
        } else if (laneHigh == 0) {
            // partial tile (Mbase==96): only rows 96..99 valid
            float4 lo;
            lo.x = acc[0]; lo.y = acc[1]; lo.z = acc[2]; lo.w = acc[3];
            *(float4*)dst = lo;
        }
    }
}

// ---------------------------------------------------------------------------
// Kernel 3: fused separable 4D conv (1 -> 16 channels -> 1), in-place on corr.
// One block per q = b*50+n.  cin + per-channel t_k live in LDS (80 KB);
// conv2 accumulators in registers.  SAME zero padding, OIHW correlation.
// ---------------------------------------------------------------------------
__global__ __launch_bounds__(256) void conv4d_fused_kernel(
    float* __restrict__ corr,
    const float* __restrict__ w1_hw, const float* __restrict__ w1_uv,
    const float* __restrict__ w2_hw, const float* __restrict__ w2_uv)
{
    __shared__ float cin[10000];   // [ij][hw] = [u*10+v][h*10+w]
    __shared__ float tk[10000];
    __shared__ float wts[576];     // w1_hw | w1_uv | w2_hw | w2_uv (16*9 each)
    const int q   = blockIdx.x;
    const int tid = threadIdx.x;
    float* base = corr + (size_t)q * 10000;

    for (int i = tid; i < 10000; i += 256) cin[i] = base[i];
    if (tid < 144) {
        wts[tid]       = w1_hw[tid];
        wts[144 + tid] = w1_uv[tid];
        wts[288 + tid] = w2_hw[tid];
        wts[432 + tid] = w2_uv[tid];
    }
    float acc[40];
    #pragma unroll
    for (int p = 0; p < 40; ++p) acc[p] = 0.f;
    __syncthreads();

    for (int k = 0; k < 16; ++k) {
        // conv1: t_k = relu( conv_hw(cin) + conv_uv(cin) )
        for (int idx = tid; idx < 10000; idx += 256) {
            const int u = idx / 1000, v = (idx / 100) % 10;
            const int h = (idx / 10) % 10, w = idx % 10;
            float s = 0.f;
            #pragma unroll
            for (int dh = -1; dh <= 1; ++dh)
                #pragma unroll
                for (int dw = -1; dw <= 1; ++dw) {
                    const int hh = h + dh, ww = w + dw;
                    if (hh >= 0 && hh < 10 && ww >= 0 && ww < 10)
                        s += wts[k*9 + (dh+1)*3 + (dw+1)] * cin[(u*10+v)*100 + hh*10 + ww];
                }
            #pragma unroll
            for (int du = -1; du <= 1; ++du)
                #pragma unroll
                for (int dv = -1; dv <= 1; ++dv) {
                    const int uu = u + du, vv = v + dv;
                    if (uu >= 0 && uu < 10 && vv >= 0 && vv < 10)
                        s += wts[144 + k*9 + (du+1)*3 + (dv+1)] * cin[(uu*10+vv)*100 + h*10 + w];
                }
            tk[idx] = fmaxf(s, 0.f);
        }
        __syncthreads();
        // conv2: accumulate channel k contribution
        int p = 0;
        for (int idx = tid; idx < 10000; idx += 256, ++p) {
            const int u = idx / 1000, v = (idx / 100) % 10;
            const int h = (idx / 10) % 10, w = idx % 10;
            float s = 0.f;
            #pragma unroll
            for (int dh = -1; dh <= 1; ++dh)
                #pragma unroll
                for (int dw = -1; dw <= 1; ++dw) {
                    const int hh = h + dh, ww = w + dw;
                    if (hh >= 0 && hh < 10 && ww >= 0 && ww < 10)
                        s += wts[288 + k*9 + (dh+1)*3 + (dw+1)] * tk[(u*10+v)*100 + hh*10 + ww];
                }
            #pragma unroll
            for (int du = -1; du <= 1; ++du)
                #pragma unroll
                for (int dv = -1; dv <= 1; ++dv) {
                    const int uu = u + du, vv = v + dv;
                    if (uu >= 0 && uu < 10 && vv >= 0 && vv < 10)
                        s += wts[432 + k*9 + (du+1)*3 + (dv+1)] * tk[(uu*10+vv)*100 + h*10 + w];
                }
            acc[p] += s;
        }
        __syncthreads();   // protect tk before next channel overwrites it
    }
    int p = 0;
    for (int idx = tid; idx < 10000; idx += 256, ++p) base[idx] = acc[p];
}

// ---------------------------------------------------------------------------
// Kernel 4: memory attention.  Per (b,n): gaussian-normalize over ij (ddof=1),
// /TEMP, softmax over ij per hw-column, sum over hw -> attn[ij]; then
// att_mem[b,n,c] = sum_ij attn[ij]*feat_m[n,c,ij] / 100.
// Deterministic (no float atomics): probabilities staged in LDS.
// ---------------------------------------------------------------------------
__global__ __launch_bounds__(128) void attn_kernel(
    const float* __restrict__ corr, const float* __restrict__ feat_m,
    float* __restrict__ att_mem)
{
    __shared__ float P[10000];     // [ij][hw] softmax probabilities
    __shared__ float attn[HW_];
    const int bn = blockIdx.x;
    const int b = bn / N_, n = bn % N_;
    const float* base = corr + (size_t)bn * 10000;   // [ij][hw]
    const int col = threadIdx.x;

    if (col < HW_) {
        float s = 0.f, s2 = 0.f, mx = -1e30f;
        for (int ij = 0; ij < HW_; ++ij) {
            const float x = base[ij * HW_ + col];
            s += x; s2 += x * x; mx = fmaxf(mx, x);
        }
        const float m   = s * 0.01f;
        const float var = fmaxf((s2 - 100.f * m * m) * (1.0f / 99.0f), 0.f);
        const float inv = 1.0f / (sqrtf(var + 1e-5f) * TEMP_);
        const float zmax = (mx - m) * inv;
        float se = 0.f;
        for (int ij = 0; ij < HW_; ++ij) {
            const float z = (base[ij * HW_ + col] - m) * inv;
            const float e = expf(z - zmax);
            P[ij * HW_ + col] = e;
            se += e;
        }
        const float rs = 1.0f / se;
        for (int ij = 0; ij < HW_; ++ij) P[ij * HW_ + col] *= rs;
    }
    __syncthreads();
    if (col < HW_) {
        float a = 0.f;
        for (int hw = 0; hw < HW_; ++hw) a += P[col * HW_ + hw];
        attn[col] = a;
    }
    __syncthreads();
    for (int c = threadIdx.x; c < C_; c += 128) {
        const float* fm = feat_m + ((size_t)n * C_ + c) * HW_;
        float s = 0.f;
        for (int ij = 0; ij < HW_; ++ij) s += attn[ij] * fm[ij];
        att_mem[((size_t)b * N_ + n) * C_ + c] = s * 0.01f;
    }
}

// ---------------------------------------------------------------------------
// Kernel 5: encoder_output[b,c] = mean over h,w of feat_b
// ---------------------------------------------------------------------------
__global__ __launch_bounds__(256) void enc_kernel(
    const float* __restrict__ feat_b, float* __restrict__ enc)
{
    const int i = blockIdx.x * blockDim.x + threadIdx.x;
    if (i < B_ * C_) {
        const float* p = feat_b + (size_t)i * HW_;
        float s = 0.f;
        for (int k = 0; k < HW_; ++k) s += p[k];
        enc[i] = s * 0.01f;
    }
}

// ---------------------------------------------------------------------------
// Kernel 6: cosine-sim content attention over memory + concat-linear head.
// One block per b.
// ---------------------------------------------------------------------------
__global__ __launch_bounds__(256) void head_kernel(
    const float* __restrict__ enc, const float* __restrict__ att_mem,
    const float* __restrict__ W_fc, const float* __restrict__ b_fc,
    float* __restrict__ out)
{
    __shared__ float e_s[C_];
    __shared__ float mv[C_];
    __shared__ float w_s[N_];
    __shared__ float sims[N_];
    __shared__ float redA[256], redB[256];
    __shared__ float enorm_s;
    const int b = blockIdx.x, tid = threadIdx.x;

    for (int c = tid; c < C_; c += 256) e_s[c] = enc[(size_t)b * C_ + c];
    __syncthreads();

    float pe = 0.f;
    for (int c = tid; c < C_; c += 256) pe += e_s[c] * e_s[c];
    redA[tid] = pe; __syncthreads();
    #pragma unroll
    for (int st = 128; st > 0; st >>= 1) {
        if (tid < st) redA[tid] += redA[tid + st];
        __syncthreads();
    }
    if (tid == 0) enorm_s = fmaxf(sqrtf(redA[0]), 1e-8f);
    __syncthreads();

    for (int n = 0; n < N_; ++n) {
        const float* am = att_mem + ((size_t)b * N_ + n) * C_;
        float pd = 0.f, pm = 0.f;
        for (int c = tid; c < C_; c += 256) {
            const float a = am[c];
            pd += e_s[c] * a; pm += a * a;
        }
        redA[tid] = pd; redB[tid] = pm; __syncthreads();
        #pragma unroll
        for (int st = 128; st > 0; st >>= 1) {
            if (tid < st) { redA[tid] += redA[tid + st]; redB[tid] += redB[tid + st]; }
            __syncthreads();
        }
        if (tid == 0)
            sims[n] = redA[0] / (enorm_s * fmaxf(sqrtf(redB[0]), 1e-8f));
        __syncthreads();
    }
    if (tid == 0) {
        float mx = -1e30f;
        for (int n = 0; n < N_; ++n) mx = fmaxf(mx, sims[n]);
        float ss = 0.f;
        for (int n = 0; n < N_; ++n) { const float e = expf(sims[n] - mx); w_s[n] = e; ss += e; }
        const float r = 1.0f / ss;
        for (int n = 0; n < N_; ++n) w_s[n] *= r;
    }
    __syncthreads();
    for (int c = tid; c < C_; c += 256) {
        float s = 0.f;
        for (int n = 0; n < N_; ++n) s += w_s[n] * att_mem[((size_t)b * N_ + n) * C_ + c];
        mv[c] = s;
    }
    __syncthreads();
    for (int j = tid; j < 1000; j += 256) {
        float s = b_fc[j];
        for (int c = 0; c < C_; ++c)  s += e_s[c] * W_fc[(size_t)c * 1000 + j];
        for (int c = 0; c < C_; ++c)  s += mv[c]  * W_fc[(size_t)(C_ + c) * 1000 + j];
        out[(size_t)b * 1000 + j] = s;
    }
}

// ---------------------------------------------------------------------------
// Host launcher.  Workspace layout (bytes, 256-aligned):
//   [0,        102400)  fb16   8*100*64  f16
//   [102400,   742400)  fm16   50*100*64 f16
//   [742400, 16742400)  corr   8*50*100*100 f32   (updated in place by conv)
//   [16742400,17766400) att_mem 8*50*640 f32
//   [17766400,17786880) enc     8*640    f32
// ---------------------------------------------------------------------------
extern "C" void kernel_launch(void* const* d_in, const int* in_sizes, int n_in,
                              void* d_out, int out_size, void* d_ws, size_t ws_size,
                              hipStream_t stream)
{
    const float* feat_b   = (const float*)d_in[0];
    const float* feat_m   = (const float*)d_in[1];
    const float* w1x1     = (const float*)d_in[2];
    const float* bn_scale = (const float*)d_in[3];
    const float* bn_bias  = (const float*)d_in[4];
    const float* w1_hw    = (const float*)d_in[5];
    const float* w1_uv    = (const float*)d_in[6];
    const float* w2_hw    = (const float*)d_in[7];
    const float* w2_uv    = (const float*)d_in[8];
    const float* W_fc     = (const float*)d_in[9];
    const float* b_fc     = (const float*)d_in[10];
    float* out = (float*)d_out;

    char* ws = (char*)d_ws;
    half_t* fb16   = (half_t*)(ws + 0);
    half_t* fm16   = (half_t*)(ws + 102400);
    float*  corr   = (float*)(ws + 742400);
    float*  attm   = (float*)(ws + 16742400);
    float*  enc    = (float*)(ws + 17766400);

    embed_kernel<<<B_*HW_ + N_*HW_, 64, 0, stream>>>(
        feat_b, feat_m, w1x1, bn_scale, bn_bias, fb16, fm16);

    corr_wmma_kernel<<<dim3(40, 7, B_), 256, 0, stream>>>(fb16, fm16, corr);

    conv4d_fused_kernel<<<B_*N_, 256, 0, stream>>>(corr, w1_hw, w1_uv, w2_hw, w2_uv);

    attn_kernel<<<B_*N_, 128, 0, stream>>>(corr, feat_m, attm);

    enc_kernel<<<(B_*C_ + 255)/256, 256, 0, stream>>>(feat_b, enc);

    head_kernel<<<B_, 256, 0, stream>>>(enc, attm, W_fc, b_fc, out);
}
